// BinarizedModelPRIMO_1726576853664
// MI455X (gfx1250) — 8-bit matrix operand order repaired and checked by reading, compiled, not run on hardware
//
#include <hip/hip_runtime.h>
#include <hip/hip_bf16.h>
#include <math.h>


typedef __attribute__((ext_vector_type(8)))  int   v8i;
typedef __attribute__((ext_vector_type(16))) int   v16i;
typedef __attribute__((ext_vector_type(8)))  float v8f;

#define FP8_P1 ((unsigned char)0x38)
#define FP8_M1 ((unsigned char)0xB8)

__device__ __forceinline__ size_t a_off(int m, int k, int Kt2  ) {
  int tile = (m >> 4) * Kt2 + (k >> 7);
  int mi = m & 15, ki = k & 63, sub = (k >> 6) & 1;
  int lane = mi + (((ki >> 3) & 1) << 4);
  int byt  = ((ki >> 4) << 3) | (ki & 7);
  return ((size_t)tile << 11) + ((size_t)lane << 6) + (sub << 5) + byt;
}
__device__ __forceinline__ size_t b_off(int k, int n, int Kt2  ) {
  int tile = (n >> 4) * Kt2 + (k >> 7);
  int ni = n & 15, ki = k & 63, sub = (k >> 6) & 1;
  int lane = ni + (((ki >> 3) & 1) << 4);
  int byt  = ((ki >> 4) << 3) | (ki & 7);
  return ((size_t)tile << 11) + ((size_t)lane << 6) + (sub << 5) + byt;
}

typedef __attribute__((address_space(3))) char lds_char;
__device__ __forceinline__ void async_cp64(void* lds_dst, const v16i* gsrc) {
  lds_char* l = (lds_char*)lds_dst;
  asm volatile(
      "global_load_async_to_lds_b128 %0, %1, off\n\t"
      "global_load_async_to_lds_b128 %0, %1, off offset:16\n\t"
      "global_load_async_to_lds_b128 %0, %1, off offset:32\n\t"
      "global_load_async_to_lds_b128 %0, %1, off offset:48"
      :: "v"(l), "v"(gsrc) : "memory");
}
__device__ __forceinline__ void wait_async0() {
  asm volatile("s_wait_asynccnt 0x0" ::: "memory");
}

__global__ void __launch_bounds__(256) pack_x_kernel(
    const float* __restrict__ x, unsigned char* __restrict__ A, int logK, int Kt2) {
  size_t idx = (size_t)blockIdx.x * 256 + threadIdx.x;
  int K = 1 << logK;
  int m = (int)(idx >> logK), k = (int)(idx & (K - 1));
  A[a_off(m, k, Kt2)] = (x[idx] >= 0.5f) ? FP8_P1 : FP8_M1;
}

__global__ void __launch_bounds__(256) pack_w_kernel(
    const float* __restrict__ W, unsigned char* __restrict__ Bp, int logN, int Kt2) {
  size_t idx = (size_t)blockIdx.x * 256 + threadIdx.x;
  int N = 1 << logN;
  int k = (int)(idx >> logN), n = (int)(idx & (N - 1));
  Bp[b_off(k, n, Kt2)] = (W[idx] >= 0.0f) ? FP8_P1 : FP8_M1;
}

__global__ void __launch_bounds__(256) gemm_bin_fp8(
    const v16i* __restrict__ A, const v16i* __restrict__ B,
    short* __restrict__ S, int N, int Kt2) {
  __shared__ v16i sA[2][256];
  __shared__ v16i sB[2][256];

  const int t    = (int)threadIdx.x;
  const int lane = t & 31;
  const int w    = t >> 5;
  const int wm   = w >> 2;
  const int wn   = w & 3;
  const int mt0  = blockIdx.y * 8;
  const int nt0  = blockIdx.x * 8;

  const int ctile = t >> 5, cl = t & 31;
  const v16i* gA = A + ((size_t)(mt0 + ctile) * Kt2) * 32 + cl;
  const v16i* gB = B + ((size_t)(nt0 + ctile) * Kt2) * 32 + cl;

  v8f acc[4][2] = {};

  async_cp64(&sA[0][t], gA);
  async_cp64(&sB[0][t], gB);
  wait_async0();
  __syncthreads();

  int p = 0;
  for (int kk = 0; kk < Kt2; ++kk) {
    if (kk + 1 < Kt2) {
      async_cp64(&sA[1 - p][t], gA + (size_t)(kk + 1) * 32);
      async_cp64(&sB[1 - p][t], gB + (size_t)(kk + 1) * 32);
    }
    v16i vb0 = sB[p][(wn * 2 + 0) * 32 + lane];
    v16i vb1 = sB[p][(wn * 2 + 1) * 32 + lane];
    v16i va0 = sA[p][(wm * 4 + 0) * 32 + lane];
    v16i va1 = sA[p][(wm * 4 + 1) * 32 + lane];
    v16i va2 = sA[p][(wm * 4 + 2) * 32 + lane];
    v16i va3 = sA[p][(wm * 4 + 3) * 32 + lane];
    acc[0][0] = __builtin_amdgcn_wmma_f32_16x16x128_fp8_fp8(va0, vb0, (short)0, acc[0][0], false, false);
    acc[0][1] = __builtin_amdgcn_wmma_f32_16x16x128_fp8_fp8(va0, vb1, (short)0, acc[0][1], false, false);
    acc[1][0] = __builtin_amdgcn_wmma_f32_16x16x128_fp8_fp8(va1, vb0, (short)0, acc[1][0], false, false);
    acc[1][1] = __builtin_amdgcn_wmma_f32_16x16x128_fp8_fp8(va1, vb1, (short)0, acc[1][1], false, false);
    acc[2][0] = __builtin_amdgcn_wmma_f32_16x16x128_fp8_fp8(va2, vb0, (short)0, acc[2][0], false, false);
    acc[2][1] = __builtin_amdgcn_wmma_f32_16x16x128_fp8_fp8(va2, vb1, (short)0, acc[2][1], false, false);
    acc[3][0] = __builtin_amdgcn_wmma_f32_16x16x128_fp8_fp8(va3, vb0, (short)0, acc[3][0], false, false);
    acc[3][1] = __builtin_amdgcn_wmma_f32_16x16x128_fp8_fp8(va3, vb1, (short)0, acc[3][1], false, false);
    wait_async0();
    __syncthreads();
    p ^= 1;
  }

  const int nlo = lane & 15, mhi = (lane >> 4) << 3;
#pragma unroll
  for (int i = 0; i < 4; ++i) {
    int r0 = (mt0 + wm * 4 + i) * 16 + mhi;
#pragma unroll
    for (int j = 0; j < 2; ++j) {
      int col = (nt0 + wn * 2 + j) * 16 + nlo;
#pragma unroll
      for (int v = 0; v < 8; ++v) {
        S[(size_t)(r0 + v) * N + col] = (short)acc[i][j][v];
      }
    }
  }
}

__global__ void __launch_bounds__(256) zero_f(float* __restrict__ p, int n) {
  int i = blockIdx.x * 256 + threadIdx.x;
  if (i < n) p[i] = 0.0f;
}

__global__ void __launch_bounds__(256) col_partial(
    const short* __restrict__ S, float* __restrict__ stats, int N, int chunk) {
  int c = blockIdx.x * 256 + threadIdx.x;
  int r0 = blockIdx.y * chunk;
  float s = 0.0f, ss = 0.0f;
  for (int i = 0; i < chunk; ++i) {
    float v = (float)S[(size_t)(r0 + i) * N + c];
    s += v; ss += v * v;
  }
  atomicAdd(&stats[2 * c], s);
  atomicAdd(&stats[2 * c + 1], ss);
}

__global__ void __launch_bounds__(256) col_finalize(float* __restrict__ stats, int N, int Bn) {
  int c = blockIdx.x * 256 + threadIdx.x;
  if (c >= N) return;
  float invB = 1.0f / (float)Bn;
  float mean = stats[2 * c] * invB;
  float var  = stats[2 * c + 1] * invB - mean * mean;
  stats[2 * c] = mean;
  stats[2 * c + 1] = rsqrtf(var + 1e-5f);
}

__global__ void __launch_bounds__(256) bn_bin_kernel(
    const short* __restrict__ S, const float* __restrict__ stats,
    const float* __restrict__ gamma, const float* __restrict__ beta, int layer,
    unsigned char* __restrict__ An, int logN, int Kt2) {
  size_t idx = (size_t)blockIdx.x * 256 + threadIdx.x;
  int N = 1 << logN;
  int r = (int)(idx >> logN), c = (int)(idx & (N - 1));
  float v = ((float)S[idx] - stats[2 * c]) * stats[2 * c + 1] * gamma[layer] + beta[layer];
  An[a_off(r, c, Kt2)] = (v >= 0.0f) ? FP8_P1 : FP8_M1;
}

__global__ void __launch_bounds__(256) bn_out_kernel(
    const short* __restrict__ S, const float* __restrict__ stats,
    const float* __restrict__ gamma, const float* __restrict__ beta, int layer,
    float* __restrict__ out, int logN) {
  size_t idx = (size_t)blockIdx.x * 256 + threadIdx.x;
  int N = 1 << logN;
  int c = (int)(idx & (N - 1));
  out[idx] = ((float)S[idx] - stats[2 * c]) * stats[2 * c + 1] * gamma[layer] + beta[layer];
}

__global__ void __launch_bounds__(256) softmax_part_kernel(
    const float* __restrict__ a, float* __restrict__ part, int N, int chunk) {
  int c = blockIdx.x * 256 + threadIdx.x;
  int r0 = blockIdx.y * chunk;
  float m = -3.4e38f, s = 0.0f;
  for (int i = 0; i < chunk; ++i) {
    float v = a[(size_t)(r0 + i) * N + c];
    float mn = fmaxf(m, v);
    s = s * __expf(m - mn) + __expf(v - mn);
    m = mn;
  }
  size_t o = ((size_t)blockIdx.y * N + c) * 2;
  part[o] = m; part[o + 1] = s;
}

__global__ void __launch_bounds__(256) softmax_combine_kernel(
    const float* __restrict__ part, float* __restrict__ sm, int N, int nch) {
  int c = blockIdx.x * 256 + threadIdx.x;
  float m = -3.4e38f, s = 0.0f;
  for (int j = 0; j < nch; ++j) {
    size_t o = ((size_t)j * N + c) * 2;
    float mj = part[o], sj = part[o + 1];
    float mn = fmaxf(m, mj);
    s = s * __expf(m - mn) + sj * __expf(mj - mn);
    m = mn;
  }
  sm[2 * c] = m;
  sm[2 * c + 1] = 1.0f / s;
}

__global__ void __launch_bounds__(256) softmax_apply_kernel(
    float* __restrict__ out, const float* __restrict__ sm, int logN) {
  size_t idx = (size_t)blockIdx.x * 256 + threadIdx.x;
  int c = (int)(idx & ((1 << logN) - 1));
  out[idx] = __expf(out[idx] - sm[2 * c]) * sm[2 * c + 1];
}

extern "C" void kernel_launch(void* const* d_in, const int* in_sizes, int n_in,
                              void* d_out, int out_size, void* d_ws, size_t ws_size,
                              hipStream_t stream) {
  (void)in_sizes; (void)n_in; (void)out_size; (void)ws_size;
  const float* x     = (const float*)d_in[0];
  const float* W0    = (const float*)d_in[1];
  const float* W1    = (const float*)d_in[2];
  const float* W2    = (const float*)d_in[3];
  const float* gamma = (const float*)d_in[4];
  const float* beta  = (const float*)d_in[5];
  float* out = (float*)d_out;

  const int B = 8192;
  const int D0 = 4096, D1 = 4096, D2 = 4096, D3 = 1024;
  const int KT2 = 4096 / 128;

  char* ws = (char*)d_ws;
  unsigned char* A0 = (unsigned char*)(ws);
  unsigned char* A1 = (unsigned char*)(ws + (size_t)33554432);
  unsigned char* Wb = (unsigned char*)(ws + (size_t)67108864);
  short*  Sb        = (short*)(ws + (size_t)83886080);
  float*  stats     = (float*)(ws + (size_t)150994944);
  float*  sm        = (float*)(ws + (size_t)150994944 + 32768);
  float*  part      = (float*)(ws + (size_t)150994944 + 32768 + 8192);

  pack_x_kernel<<<(B * D0) / 256, 256, 0, stream>>>(x, A0, 12, KT2);
  pack_w_kernel<<<(D0 * D1) / 256, 256, 0, stream>>>(W0, Wb, 12, KT2);
  gemm_bin_fp8<<<dim3(D1 / 128, B / 128), 256, 0, stream>>>(
      (const v16i*)A0, (const v16i*)Wb, Sb, D1, KT2);
  zero_f<<<(2 * D1 + 255) / 256, 256, 0, stream>>>(stats, 2 * D1);
  col_partial<<<dim3(D1 / 256, B / 256), 256, 0, stream>>>(Sb, stats, D1, 256);
  col_finalize<<<(D1 + 255) / 256, 256, 0, stream>>>(stats, D1, B);
  bn_bin_kernel<<<(B * D1) / 256, 256, 0, stream>>>(Sb, stats, gamma, beta, 0, A1, 12, KT2);

  pack_w_kernel<<<(D1 * D2) / 256, 256, 0, stream>>>(W1, Wb, 12, KT2);
  gemm_bin_fp8<<<dim3(D2 / 128, B / 128), 256, 0, stream>>>(
      (const v16i*)A1, (const v16i*)Wb, Sb, D2, KT2);
  zero_f<<<(2 * D2 + 255) / 256, 256, 0, stream>>>(stats, 2 * D2);
  col_partial<<<dim3(D2 / 256, B / 256), 256, 0, stream>>>(Sb, stats, D2, 256);
  col_finalize<<<(D2 + 255) / 256, 256, 0, stream>>>(stats, D2, B);
  bn_bin_kernel<<<(B * D2) / 256, 256, 0, stream>>>(Sb, stats, gamma, beta, 1, A0, 12, KT2);

  pack_w_kernel<<<(D2 * D3) / 256, 256, 0, stream>>>(W2, Wb, 10, KT2);
  gemm_bin_fp8<<<dim3(D3 / 128, B / 128), 256, 0, stream>>>(
      (const v16i*)A0, (const v16i*)Wb, Sb, D3, KT2);
  zero_f<<<(2 * D3 + 255) / 256, 256, 0, stream>>>(stats, 2 * D3);
  col_partial<<<dim3(D3 / 256, B / 256), 256, 0, stream>>>(Sb, stats, D3, 256);
  col_finalize<<<(D3 + 255) / 256, 256, 0, stream>>>(stats, D3, B);
  bn_out_kernel<<<(B * D3) / 256, 256, 0, stream>>>(Sb, stats, gamma, beta, 2, out, 10);

  softmax_part_kernel<<<dim3(D3 / 256, 32), 256, 0, stream>>>(out, part, D3, B / 32);
  softmax_combine_kernel<<<D3 / 256, 256, 0, stream>>>(part, sm, D3, 32);
  softmax_apply_kernel<<<(B * D3) / 256, 256, 0, stream>>>(out, sm, 10);
}
